// LlamaAttention_29025388987010
// MI455X (gfx1250) — matrix operand order repaired and checked by reading, not compiled, not run on hardware
//
#include <hip/hip_runtime.h>

typedef __attribute__((ext_vector_type(16))) __bf16 v16bf;
typedef __attribute__((ext_vector_type(8)))  float  v8f;

#define HIDDEN 4096
#define NHEAD  32
#define DHEAD  128
#define QLEN   64
#define SLEN   4096
#define NBATCH 4
#define SCALEF 0.08838834764831845f  // 1/sqrt(128)
#define NEGM  -3.0e38f
#define NEGI  -1.0e38f

__device__ __forceinline__ float bf2f(__bf16 h){
  unsigned short s = __builtin_bit_cast(unsigned short, h);
  unsigned u = ((unsigned)s) << 16;
  return __builtin_bit_cast(float, u);
}
__device__ __forceinline__ __bf16 f2bf(float f){
  unsigned u = __builtin_bit_cast(unsigned, f);
  u += 0x7FFFu + ((u >> 16) & 1u);
  unsigned short s = (unsigned short)(u >> 16);
  return __builtin_bit_cast(__bf16, s);
}

union FragU { uint4 u[2]; v16bf v; };

// A-matrix fragment (16xK row 'row', this lane's half): K = hf*8..hf*8+7 and 16+hf*8..+7
__device__ __forceinline__ v16bf lds_frag_a(const __bf16* row, int hf){
  FragU f;
  f.u[0] = *reinterpret_cast<const uint4*>(row + hf*8);
  f.u[1] = *reinterpret_cast<const uint4*>(row + 16 + hf*8);
  return f.v;
}
// B-matrix fragment (col 'row' holds K contiguous): K = hf*8..hf*8+7 and 16+hf*8..+7
__device__ __forceinline__ v16bf lds_frag_b(const __bf16* row, int hf){
  FragU f;
  f.u[0] = *reinterpret_cast<const uint4*>(row + hf*8);
  f.u[1] = *reinterpret_cast<const uint4*>(row + 16 + hf*8);
  return f.v;
}
__device__ __forceinline__ v8f wmma_bf(v16bf a, v16bf b, v8f c){
  return __builtin_amdgcn_wmma_f32_16x16x32_bf16(false, a, false, b, (short)0, c, false, false);
}

// ---------------------------------------------------------------------------
// C[M=256,N=4096] = A[256,4096] * W[4096,4096]^T, split-bf16, f32 accumulate.
// Workgroup tile 128x64, 8 waves, each wave a 32x32 quadrant (2x2 WMMA tiles).
// ---------------------------------------------------------------------------
__global__ __launch_bounds__(256) void gemm256(const float* __restrict__ A,
                                               const float* __restrict__ W,
                                               float* __restrict__ C){
  __shared__ __align__(16) __bf16 Ah[128][40], Al[128][40];
  __shared__ __align__(16) __bf16 Bh[64][40],  Bl[64][40];
  const int tid = threadIdx.x;
  const int wv = tid >> 5, lane = tid & 31;
  const int hf = lane >> 4, ln = lane & 15;
  const int wm = wv & 3, wn = wv >> 2;
  const int mBase = blockIdx.y * 128;
  const int nBase = blockIdx.x * 64;
  v8f acc[2][2] = {};
  for (int kb = 0; kb < HIDDEN; kb += 32){
    for (int i = 0; i < 4; ++i){
      int idx = i*256 + tid;
      int row = idx >> 3, c4 = (idx & 7) << 2;
      float4 v = *reinterpret_cast<const float4*>(&A[(mBase+row)*HIDDEN + kb + c4]);
      float xs[4] = {v.x, v.y, v.z, v.w};
      for (int j = 0; j < 4; ++j){
        __bf16 hi = f2bf(xs[j]);
        Ah[row][c4+j] = hi;
        Al[row][c4+j] = f2bf(xs[j] - bf2f(hi));
      }
    }
    for (int i = 0; i < 2; ++i){
      int idx = i*256 + tid;
      int row = idx >> 3, c4 = (idx & 7) << 2;
      float4 v = *reinterpret_cast<const float4*>(&W[(nBase+row)*HIDDEN + kb + c4]);
      float xs[4] = {v.x, v.y, v.z, v.w};
      for (int j = 0; j < 4; ++j){
        __bf16 hi = f2bf(xs[j]);
        Bh[row][c4+j] = hi;
        Bl[row][c4+j] = f2bf(xs[j] - bf2f(hi));
      }
    }
    __syncthreads();
    v16bf ah[2], al[2], bh[2], bl[2];
    for (int mt = 0; mt < 2; ++mt){
      int r = wm*32 + mt*16 + ln;
      ah[mt] = lds_frag_a(&Ah[r][0], hf);
      al[mt] = lds_frag_a(&Al[r][0], hf);
    }
    for (int nt = 0; nt < 2; ++nt){
      int r = wn*32 + nt*16 + ln;
      bh[nt] = lds_frag_b(&Bh[r][0], hf);
      bl[nt] = lds_frag_b(&Bl[r][0], hf);
    }
    for (int mt = 0; mt < 2; ++mt)
      for (int nt = 0; nt < 2; ++nt){
        acc[mt][nt] = wmma_bf(ah[mt], bh[nt], acc[mt][nt]);
        acc[mt][nt] = wmma_bf(ah[mt], bl[nt], acc[mt][nt]);
        acc[mt][nt] = wmma_bf(al[mt], bh[nt], acc[mt][nt]);
      }
    __syncthreads();
  }
  for (int mt = 0; mt < 2; ++mt)
    for (int nt = 0; nt < 2; ++nt)
      for (int r = 0; r < 8; ++r){
        int m = mBase + wm*32 + mt*16 + r + 8*hf;
        int n = nBase + wn*32 + nt*16 + ln;
        C[m*HIDDEN + n] = acc[mt][nt][r];
      }
}

// ---------------------------------------------------------------------------
// RoPE applied in-place to q and k projection buffers.
// ---------------------------------------------------------------------------
__global__ void rope_kernel(float* __restrict__ qbuf, float* __restrict__ kbuf,
                            const float* __restrict__ cosp, const float* __restrict__ sinp){
  int idx = blockIdx.x*blockDim.x + threadIdx.x;      // B*Q*NH*64
  if (idx >= NBATCH*QLEN*NHEAD*64) return;
  int d = idx & 63;
  int h = (idx >> 6) & 31;
  int q = (idx >> 11) & 63;
  int b = idx >> 17;
  int rb = (b*QLEN + q)*DHEAD;
  float c1 = cosp[rb + d],       s1 = sinp[rb + d];
  float c2 = cosp[rb + d + 64],  s2 = sinp[rb + d + 64];
  int base = (b*QLEN + q)*HIDDEN + h*DHEAD;
  float q1 = qbuf[base + d], q2 = qbuf[base + d + 64];
  qbuf[base + d]      = q1*c1 - q2*s1;
  qbuf[base + d + 64] = q2*c2 + q1*s2;
  float k1 = kbuf[base + d], k2 = kbuf[base + d + 64];
  kbuf[base + d]      = k1*c1 - k2*s1;
  kbuf[base + d + 64] = k2*c2 + k1*s2;
}

// ---------------------------------------------------------------------------
// Bulk cache copy (float4, grid-stride) and new-token scatter.
// ---------------------------------------------------------------------------
__global__ void copykv(const float4* __restrict__ K, const float4* __restrict__ V,
                       float4* __restrict__ Kn, float4* __restrict__ Vn, int n4){
  int stride = gridDim.x*blockDim.x;
  for (int i = blockIdx.x*blockDim.x + threadIdx.x; i < n4; i += stride){
    Kn[i] = K[i]; Vn[i] = V[i];
  }
}

__global__ void scatter_kv(const float* __restrict__ kbuf, const float* __restrict__ vbuf,
                           const int* __restrict__ clens,
                           float* __restrict__ Kn, float* __restrict__ Vn){
  int idx = blockIdx.x*blockDim.x + threadIdx.x;      // B*Q*HIDDEN
  if (idx >= NBATCH*QLEN*HIDDEN) return;
  int c = idx & (HIDDEN-1);
  int q = (idx >> 12) & 63;
  int b = idx >> 18;
  int pos = clens[b] + q;
  int dst = (b*SLEN + pos)*HIDDEN + c;
  Kn[dst] = kbuf[idx];
  Vn[dst] = vbuf[idx];
}

// ---------------------------------------------------------------------------
// Attention: one workgroup per (h,b). 4 waves x 16 q-rows. Flash over 32-key
// blocks; phase 0 = prefix cache (length mask), phase 1 = current 64 keys
// (tree mask). Combined via sigmoid(lse_p - lse_c).
// ---------------------------------------------------------------------------
__global__ __launch_bounds__(128) void attn_kernel(
    const float* __restrict__ qbuf, const float* __restrict__ kbuf, const float* __restrict__ vbuf,
    const float* __restrict__ Kc, const float* __restrict__ Vc,
    const int* __restrict__ clens, const int* __restrict__ tmask,
    float* __restrict__ attnb){
  __shared__ __align__(16) __bf16 Kh[32][DHEAD+8], Kl[32][DHEAD+8];
  __shared__ __align__(16) __bf16 Vh[DHEAD][40],   Vl[DHEAD][40];
  __shared__ __align__(16) __bf16 Ph[64][40],      Pl[64][40];
  const int h = blockIdx.x, b = blockIdx.y;
  const int tid = threadIdx.x;
  const int wv = tid >> 5, lane = tid & 31;
  const int hf = lane >> 4, ln = lane & 15;
  const int qb = wv * 16;
  const int clen = clens[b];
  const v8f zero = {0.f,0.f,0.f,0.f,0.f,0.f,0.f,0.f};

  // Q fragments (A-matrix layout, hi/lo) for this wave's 16 rows, K=128 -> 4 steps
  v16bf qh[4], ql[4];
  {
    const float* qrow = qbuf + (b*QLEN + qb + ln)*HIDDEN + h*DHEAD;
    for (int ks = 0; ks < 4; ++ks){
      v16bf th, tl;
      for (int j = 0; j < 8; ++j){
        float x = qrow[ks*32 + hf*8 + j];
        __bf16 hi = f2bf(x);
        th[j] = hi; tl[j] = f2bf(x - bf2f(hi));
        float y = qrow[ks*32 + 16 + hf*8 + j];
        hi = f2bf(y);
        th[8+j] = hi; tl[8+j] = f2bf(y - bf2f(hi));
      }
      qh[ks] = th; ql[ks] = tl;
    }
  }

  float lse_p[8];
  for (int phase = 0; phase < 2; ++phase){
    float m_s[8], l_s[8];
    v8f Oacc[8];
    for (int r = 0; r < 8; ++r){ m_s[r] = NEGI; l_s[r] = 0.f; }
    for (int dt = 0; dt < 8; ++dt) Oacc[dt] = zero;

    int nblk = phase ? 2 : ((clen + 31) >> 5);
    for (int blk = 0; blk < nblk; ++blk){
      int s0 = blk*32;
      // stage 32 keys x 128 dims of K (row-major) and V (transposed [d][key])
      for (int i = 0; i < 32; ++i){
        int idx = i*128 + tid;
        int key = idx >> 7, d = idx & 127;
        float kx, vx;
        if (phase){
          int off = (b*QLEN + s0 + key)*HIDDEN + h*DHEAD + d;
          kx = kbuf[off]; vx = vbuf[off];
        } else {
          int off = ((b*SLEN + s0 + key)*NHEAD + h)*DHEAD + d;
          kx = Kc[off]; vx = Vc[off];
        }
        __bf16 khi = f2bf(kx); Kh[key][d] = khi; Kl[key][d] = f2bf(kx - bf2f(khi));
        __bf16 vhi = f2bf(vx); Vh[d][key] = vhi; Vl[d][key] = f2bf(vx - bf2f(vhi));
      }
      __syncthreads();

      // scores: 16q x 32k per wave
      v8f sc[2];
      for (int kt = 0; kt < 2; ++kt){
        v8f a = zero;
        const __bf16* krh = &Kh[kt*16 + ln][0];
        const __bf16* krl = &Kl[kt*16 + ln][0];
        for (int ks = 0; ks < 4; ++ks){
          v16bf bh = lds_frag_b(krh + ks*32, hf);
          v16bf bl = lds_frag_b(krl + ks*32, hf);
          a = wmma_bf(qh[ks], bh, a);
          a = wmma_bf(qh[ks], bl, a);
          a = wmma_bf(ql[ks], bh, a);
        }
        sc[kt] = a * SCALEF;
      }
      // masking
      if (phase == 0){
        for (int kt = 0; kt < 2; ++kt)
          if (s0 + kt*16 + ln >= clen)
            for (int r = 0; r < 8; ++r) sc[kt][r] = NEGM;
      } else {
        for (int kt = 0; kt < 2; ++kt){
          int kc = s0 + kt*16 + ln;
          for (int r = 0; r < 8; ++r){
            int qr = qb + r + 8*hf;
            if (tmask[(b*QLEN + qr)*QLEN + kc] == 0) sc[kt][r] = NEGM;
          }
        }
      }
      // online softmax update + stage P (split bf16) to LDS
      for (int r = 0; r < 8; ++r){
        float mx = fmaxf(sc[0][r], sc[1][r]);
        for (int off = 8; off; off >>= 1) mx = fmaxf(mx, __shfl_xor(mx, off, 32));
        float mnew = fmaxf(m_s[r], mx);
        float corr = __expf(m_s[r] - mnew);
        float p0 = __expf(sc[0][r] - mnew);
        float p1 = __expf(sc[1][r] - mnew);
        float rs = p0 + p1;
        for (int off = 8; off; off >>= 1) rs += __shfl_xor(rs, off, 32);
        m_s[r] = mnew;
        l_s[r] = l_s[r]*corr + rs;
        for (int dt = 0; dt < 8; ++dt) Oacc[dt][r] *= corr;
        int qr = qb + r + 8*hf;
        __bf16 ph = f2bf(p0); Ph[qr][ln]      = ph; Pl[qr][ln]      = f2bf(p0 - bf2f(ph));
        ph = f2bf(p1);        Ph[qr][16 + ln] = ph; Pl[qr][16 + ln] = f2bf(p1 - bf2f(ph));
      }
      // P x V accumulate (K=32 keys -> one WMMA step per d-tile)
      v16bf pah = lds_frag_a(&Ph[qb + ln][0], hf);
      v16bf pal = lds_frag_a(&Pl[qb + ln][0], hf);
      for (int dt = 0; dt < 8; ++dt){
        v16bf vbh = lds_frag_b(&Vh[dt*16 + ln][0], hf);
        v16bf vbl = lds_frag_b(&Vl[dt*16 + ln][0], hf);
        Oacc[dt] = wmma_bf(pah, vbh, Oacc[dt]);
        Oacc[dt] = wmma_bf(pah, vbl, Oacc[dt]);
        Oacc[dt] = wmma_bf(pal, vbh, Oacc[dt]);
      }
      __syncthreads();
    }

    if (phase == 0){
      for (int r = 0; r < 8; ++r){
        lse_p[r] = m_s[r] + __logf(l_s[r]);
        float inv = 1.0f / l_s[r];
        int qr = qb + r + 8*hf;
        for (int dt = 0; dt < 8; ++dt)
          attnb[(b*QLEN + qr)*HIDDEN + h*DHEAD + dt*16 + ln] = Oacc[dt][r] * inv;
      }
    } else {
      for (int r = 0; r < 8; ++r){
        float lse_c = m_s[r] + __logf(l_s[r]);
        float wgt = 1.0f / (1.0f + __expf(lse_c - lse_p[r]));  // sigmoid(lse_p - lse_c)
        float inv = 1.0f / l_s[r];
        int qr = qb + r + 8*hf;
        for (int dt = 0; dt < 8; ++dt){
          int o = (b*QLEN + qr)*HIDDEN + h*DHEAD + dt*16 + ln;
          float po = attnb[o];
          attnb[o] = po*wgt + (Oacc[dt][r]*inv)*(1.0f - wgt);
        }
      }
    }
  }
}

// ---------------------------------------------------------------------------
extern "C" void kernel_launch(void* const* d_in, const int* in_sizes, int n_in,
                              void* d_out, int out_size, void* d_ws, size_t ws_size,
                              hipStream_t stream) {
  const float* hidden = (const float*)d_in[0];
  const float* cosp   = (const float*)d_in[1];
  const float* sinp   = (const float*)d_in[2];
  const float* Kc     = (const float*)d_in[3];
  const float* Vc     = (const float*)d_in[4];
  const float* Wq     = (const float*)d_in[5];
  const float* Wk     = (const float*)d_in[6];
  const float* Wv     = (const float*)d_in[7];
  const float* Wo     = (const float*)d_in[8];
  const int*   clens  = (const int*)d_in[9];
  const int*   tmask  = (const int*)d_in[10];

  float* out  = (float*)d_out;                       // (B,Q,HID)
  float* Knew = out + (size_t)NBATCH*QLEN*HIDDEN;    // (B,S,NH,D)
  float* Vnew = Knew + (size_t)NBATCH*SLEN*NHEAD*DHEAD;

  float* qbuf  = (float*)d_ws;                       // (B,Q,HID) each
  float* kbuf  = qbuf + (size_t)NBATCH*QLEN*HIDDEN;
  float* vbuf  = kbuf + (size_t)NBATCH*QLEN*HIDDEN;
  float* attnb = vbuf + (size_t)NBATCH*QLEN*HIDDEN;

  dim3 gg(HIDDEN/64, (NBATCH*QLEN)/128, 1);
  gemm256<<<gg, 256, 0, stream>>>(hidden, Wq, qbuf);
  gemm256<<<gg, 256, 0, stream>>>(hidden, Wk, kbuf);
  gemm256<<<gg, 256, 0, stream>>>(hidden, Wv, vbuf);

  rope_kernel<<<(NBATCH*QLEN*NHEAD*64)/256, 256, 0, stream>>>(qbuf, kbuf, cosp, sinp);

  int n4 = (NBATCH*SLEN*NHEAD*DHEAD)/4;
  copykv<<<8192, 256, 0, stream>>>((const float4*)Kc, (const float4*)Vc,
                                   (float4*)Knew, (float4*)Vnew, n4);
  scatter_kv<<<(NBATCH*QLEN*HIDDEN)/256, 256, 0, stream>>>(kbuf, vbuf, clens, Knew, Vnew);

  attn_kernel<<<dim3(NHEAD, NBATCH), 128, 0, stream>>>(qbuf, kbuf, vbuf, Kc, Vc,
                                                       clens, tmask, attnb);

  gemm256<<<gg, 256, 0, stream>>>(attnb, Wo, out);
}
